// MetalOptimizedmLSTMBlock_45638322487832
// MI455X (gfx1250) — compile-verified, matrix operand order checked by reading, not run on hardware
//
#include <hip/hip_runtime.h>
#include <math.h>

// ---------------- problem constants ----------------
#define Dm   768
#define NH   8
#define DH   96
#define HID  768
#define Pp   1536
#define CAP  15.0f
#define EPS  1e-6f
#define Bb   8
#define Ss   2048
#define Mrows (Bb*Ss)             // 16384
#define N1   (Pp + HID)           // 2304  (xl | r)
#define ZB   (2*NH + 4*HID)       // 3088  (z width, biased)
#define N2   (ZB + HID)           // 3856  (z | skip)
#define N2P  3872                 // padded to a multiple of 32 (121 * 32)

typedef __attribute__((ext_vector_type(16))) _Float16 v16h;
typedef __attribute__((ext_vector_type(8)))  _Float16 v8h;
typedef __attribute__((ext_vector_type(8)))  float    v8f;

// explicit global-address-space pointers: keeps fragment loads as
// global_load_b128 (LOADcnt only) even through loop-carried pointer arrays,
// instead of flat_load_b128 (LOADcnt+DScnt + aperture path).
typedef const _Float16 __attribute__((address_space(1)))* gcf16p;
typedef const v8h      __attribute__((address_space(1)))* gv8hp;

// ---------------- helpers ----------------
__device__ __forceinline__ float sigmoidf_(float x) { return 1.0f / (1.0f + __expf(-x)); }
__device__ __forceinline__ float siluf_(float x)    { return x * sigmoidf_(x); }
__device__ __forceinline__ float softcapf_(float x) { return CAP * tanhf(x * (1.0f / CAP)); }

// Load a 16x32 f16 fragment (A layout per CDNA5 ISA 7.12.2) from a
// precomputed per-lane base pointer plus a compile-time element offset.
// elemOff folds into the global_load_b128 immediate offset.
__device__ __forceinline__ v16h load_frag_at(const _Float16* p, int elemOff) {
  gcf16p gp = (gcf16p)p;
  v8h lo = *(gv8hp)(gp + elemOff);        // K-half 0 (16B)
  v8h hi = *(gv8hp)(gp + elemOff + 16);   // K-half 1 (16B)
  v16h f;
#pragma unroll
  for (int i = 0; i < 8; ++i) { f[i] = lo[i]; f[i + 8] = hi[i]; }
  return f;
}

// ---------------- kernels ----------------
__global__ void cvt_f32_to_f16(const float* __restrict__ src, _Float16* __restrict__ dst, int n) {
  int i = blockIdx.x * blockDim.x + threadIdx.x;
  if (i < n) dst[i] = (_Float16)src[i];
}

__global__ void zero_f16(_Float16* __restrict__ dst, int n) {
  int i = blockIdx.x * blockDim.x + threadIdx.x;
  if (i < n) dst[i] = (_Float16)0.0f;
}

// LayerNorm over D=768, emit f16. One block per row, 256 threads * 3 elems.
__global__ void ln_kernel(const float* __restrict__ x, const float* __restrict__ w,
                          const float* __restrict__ b, _Float16* __restrict__ xn) {
  __shared__ float s1[256], s2[256];
  int row = blockIdx.x, tid = threadIdx.x;
  const float* xr = x + (size_t)row * Dm;
  float v0 = xr[tid], v1 = xr[tid + 256], v2 = xr[tid + 512];
  s1[tid] = v0 + v1 + v2;
  s2[tid] = v0 * v0 + v1 * v1 + v2 * v2;
  __syncthreads();
  for (int off = 128; off > 0; off >>= 1) {
    if (tid < off) { s1[tid] += s1[tid + off]; s2[tid] += s2[tid + off]; }
    __syncthreads();
  }
  float mu = s1[0] * (1.0f / Dm);
  float var = s2[0] * (1.0f / Dm) - mu * mu;
  float rstd = rsqrtf(var + EPS);
  _Float16* xo = xn + (size_t)row * Dm;
  xo[tid]       = (_Float16)((v0 - mu) * rstd * w[tid]       + b[tid]);
  xo[tid + 256] = (_Float16)((v1 - mu) * rstd * w[tid + 256] + b[tid + 256]);
  xo[tid + 512] = (_Float16)((v2 - mu) * rstd * w[tid + 512] + b[tid + 512]);
}

// C[M,N] = A[M,K] * B[N,K]^T via v_wmma_f32_16x16x32_f16.
// One wave owns a 64x32 C tile: 4x2 accum frags; per K-step 4 A + 2 B frags.
// Software-pipelined K loop (double-buffered frags, K % 64 == 0), pointer-
// increment addressing (constant immediate offsets within each 128B K step).
// __launch_bounds__(256, 2): ~200 VGPRs live, so allow up to 512/wave — no
// scratch spills; 2-4 waves/SIMD + the pipeline hide L2 latency fine.
__global__ void __launch_bounds__(256, 2)
gemm_wmma_f16(const _Float16* __restrict__ A, const _Float16* __restrict__ Bm,
              float* __restrict__ C, const float* __restrict__ bias, int biasN,
              const float* __restrict__ residual, int M, int N, int K) {
  int wavesPerBlock = blockDim.x >> 5;
  int w = blockIdx.x * wavesPerBlock + (threadIdx.x >> 5);
  int ntiles = N >> 5;               // 32-col tiles
  int mTile = w / ntiles;
  int nTile = w - mTile * ntiles;
  int m0 = mTile << 6;               // 64-row tile
  int n0 = nTile << 5;               // 32-col tile
  if (m0 >= M) return;

  int lane = threadIdx.x & 31;
  int lm = lane & 15;
  int lg = lane >> 4;

  // per-lane fragment base pointers (row + K-half-group offset)
  const _Float16* pa[4];
  const _Float16* pb[2];
#pragma unroll
  for (int i = 0; i < 4; ++i) pa[i] = A  + (size_t)(m0 + 16 * i + lm) * K + lg * 8;
#pragma unroll
  for (int j = 0; j < 2; ++j) pb[j] = Bm + (size_t)(n0 + 16 * j + lm) * K + lg * 8;

  v8f acc[4][2];
#pragma unroll
  for (int i = 0; i < 4; ++i)
#pragma unroll
    for (int j = 0; j < 2; ++j) acc[i][j] = (v8f){};

  v16h af[4], bf[2], naf[4], nbf[2];
#pragma unroll
  for (int i = 0; i < 4; ++i) af[i] = load_frag_at(pa[i], 0);
#pragma unroll
  for (int j = 0; j < 2; ++j) bf[j] = load_frag_at(pb[j], 0);

  int steps = K >> 6;                // K / 64, K % 64 == 0 for all our GEMMs
  for (int s = 0; s < steps - 1; ++s) {
    __builtin_prefetch(pb[0] + 128, 0, 3);
#pragma unroll
    for (int i = 0; i < 4; ++i) naf[i] = load_frag_at(pa[i], 32);   // K+32
#pragma unroll
    for (int j = 0; j < 2; ++j) nbf[j] = load_frag_at(pb[j], 32);
#pragma unroll
    for (int i = 0; i < 4; ++i)
#pragma unroll
      for (int j = 0; j < 2; ++j)
        acc[i][j] = __builtin_amdgcn_wmma_f32_16x16x32_f16(false, af[i], false, bf[j],
                                                           (short)0, acc[i][j], false, false);
#pragma unroll
    for (int i = 0; i < 4; ++i) af[i] = load_frag_at(pa[i], 64);    // K+64
#pragma unroll
    for (int j = 0; j < 2; ++j) bf[j] = load_frag_at(pb[j], 64);
#pragma unroll
    for (int i = 0; i < 4; ++i)
#pragma unroll
      for (int j = 0; j < 2; ++j)
        acc[i][j] = __builtin_amdgcn_wmma_f32_16x16x32_f16(false, naf[i], false, nbf[j],
                                                           (short)0, acc[i][j], false, false);
#pragma unroll
    for (int i = 0; i < 4; ++i) pa[i] += 64;
#pragma unroll
    for (int j = 0; j < 2; ++j) pb[j] += 64;
  }
  // peeled final 64-K step (no conditional loads in steady state)
#pragma unroll
  for (int i = 0; i < 4; ++i) naf[i] = load_frag_at(pa[i], 32);
#pragma unroll
  for (int j = 0; j < 2; ++j) nbf[j] = load_frag_at(pb[j], 32);
#pragma unroll
  for (int i = 0; i < 4; ++i)
#pragma unroll
    for (int j = 0; j < 2; ++j)
      acc[i][j] = __builtin_amdgcn_wmma_f32_16x16x32_f16(false, af[i], false, bf[j],
                                                         (short)0, acc[i][j], false, false);
#pragma unroll
  for (int i = 0; i < 4; ++i)
#pragma unroll
    for (int j = 0; j < 2; ++j)
      acc[i][j] = __builtin_amdgcn_wmma_f32_16x16x32_f16(false, naf[i], false, nbf[j],
                                                         (short)0, acc[i][j], false, false);

  int rofs = lg * 8;                 // C/D layout: row = r + 8*(lane>=16)
#pragma unroll
  for (int j = 0; j < 2; ++j) {
    int col = n0 + j * 16 + lm;
    float bv = (bias != nullptr && col < biasN) ? bias[col] : 0.0f;
#pragma unroll
    for (int i = 0; i < 4; ++i) {
#pragma unroll
      for (int r = 0; r < 8; ++r) {
        int row = m0 + i * 16 + rofs + r;
        float v = acc[i][j][r] + bv;
        if (residual != nullptr) v += residual[(size_t)row * N + col];
        C[(size_t)row * N + col] = v;
      }
    }
  }
}

// Causal conv (K=4) along the FEATURE dim + SiLU, f32 -> f16.
__global__ void conv_silu_kernel(const float* __restrict__ Y1, const float* __restrict__ cw,
                                 _Float16* __restrict__ XC) {
  int idx = blockIdx.x * blockDim.x + threadIdx.x;
  int row = idx / Pp;
  int p   = idx - row * Pp;
  const float* xl = Y1 + (size_t)row * N1;   // xl = cols [0, P) of Y1
  float w0 = cw[0], w1 = cw[1], w2 = cw[2], w3 = cw[3];
  float acc = w3 * xl[p];
  if (p >= 1) acc += w2 * xl[p - 1];
  if (p >= 2) acc += w1 * xl[p - 2];
  if (p >= 3) acc += w0 * xl[p - 3];
  XC[(size_t)row * Pp + p] = (_Float16)siluf_(acc);
}

// Matrix-state recurrence. One block per (b,h); 96 threads (3 waves).
// Thread d owns row d of c[96x96] in registers and n[d].
__global__ void __launch_bounds__(96) recurrence_kernel(const float* __restrict__ Z,
                                                        float* __restrict__ H) {
  __shared__ float qs[DH], ks[DH], red[3];
  int bh = blockIdx.x;
  int b = bh / NH, h = bh - b * NH;
  int d = threadIdx.x;
  int lane = d & 31, wid = d >> 5;
  const float inv_sqrt_dh = 0.10206207262f;  // 1/sqrt(96)

  float c[DH];
#pragma unroll
  for (int e = 0; e < DH; ++e) c[e] = 0.0f;
  float nd = 1.0f;         // n initialized to ones
  float m  = 0.0f;

  for (int t = 0; t < Ss; ++t) {
    size_t zrow = ((size_t)b * Ss + t) * (size_t)N2P;
    float i_ = softcapf_(Z[zrow + h]);
    float f_ = softcapf_(Z[zrow + NH + h]);
    int col = h * DH + d;
    float o_ = sigmoidf_(Z[zrow + 16 + col]);
    float qd = Z[zrow + 16 + HID + col];
    float kd = Z[zrow + 16 + 2 * HID + col] * inv_sqrt_dh;
    float vd = Z[zrow + 16 + 3 * HID + col];
    qs[d] = qd; ks[d] = kd;
    __syncthreads();

    float m_t = fmaxf(f_ + m, i_);
    float ig = __expf(i_ - m_t);
    float fg = __expf(f_ + m - m_t);
    m = m_t;

    float igv = ig * vd;
    float num = 0.0f;
#pragma unroll
    for (int e = 0; e < DH; ++e) {          // uniform LDS reads -> broadcast
      c[e] = fg * c[e] + igv * ks[e];
      num += c[e] * qs[e];
    }
    nd = fg * nd + ig * kd;

    float pd = nd * qd;                      // denominator reduction (96 lanes)
#pragma unroll
    for (int off = 16; off >= 1; off >>= 1) pd += __shfl_xor(pd, off, 32);
    if (lane == 0) red[wid] = pd;
    __syncthreads();
    float den = fmaxf(red[0] + red[1] + red[2], 1.0f);

    H[((size_t)b * Ss + t) * HID + col] = o_ * num / den;
    __syncthreads();                         // qs/ks reused next step
  }
}

// hid-LN over HID=768 + skip + silu(r) multiplier; emit f16 for down-proj.
__global__ void post_kernel(const float* __restrict__ H, const float* __restrict__ Z,
                            const float* __restrict__ Y1, const float* __restrict__ hw,
                            const float* __restrict__ hb, _Float16* __restrict__ U) {
  __shared__ float s1[256], s2[256];
  int row = blockIdx.x, tid = threadIdx.x;
  const float* hr = H + (size_t)row * HID;
  float v0 = hr[tid], v1 = hr[tid + 256], v2 = hr[tid + 512];
  s1[tid] = v0 + v1 + v2;
  s2[tid] = v0 * v0 + v1 * v1 + v2 * v2;
  __syncthreads();
  for (int off = 128; off > 0; off >>= 1) {
    if (tid < off) { s1[tid] += s1[tid + off]; s2[tid] += s2[tid + off]; }
    __syncthreads();
  }
  float mu = s1[0] * (1.0f / HID);
  float rstd = rsqrtf(s2[0] * (1.0f / HID) - mu * mu + EPS);
  const float* skip = Z + (size_t)row * N2P + ZB;   // cols [3088, 3856)
  const float* rr   = Y1 + (size_t)row * N1 + Pp;   // r = cols [1536, 2304)
  _Float16* uo = U + (size_t)row * HID;
#pragma unroll
  for (int it = 0; it < 3; ++it) {
    int j = tid + it * 256;
    float v = (it == 0) ? v0 : (it == 1) ? v1 : v2;
    float u = ((v - mu) * rstd * hw[j] + hb[j] + skip[j]) * siluf_(rr[j]);
    uo[j] = (_Float16)u;
  }
}

// ---------------- launch ----------------
extern "C" void kernel_launch(void* const* d_in, const int* in_sizes, int n_in,
                              void* d_out, int out_size, void* d_ws, size_t ws_size,
                              hipStream_t stream) {
  const float* x      = (const float*)d_in[0];
  const float* ln_w   = (const float*)d_in[1];
  const float* ln_b   = (const float*)d_in[2];
  const float* up_l_w = (const float*)d_in[3];
  const float* up_r_w = (const float*)d_in[4];
  const float* conv_w = (const float*)d_in[5];
  const float* skip_w = (const float*)d_in[6];
  const float* fused_w= (const float*)d_in[7];
  const float* fused_b= (const float*)d_in[8];
  const float* hid_w  = (const float*)d_in[9];
  const float* hid_b  = (const float*)d_in[10];
  const float* down_w = (const float*)d_in[11];
  float* out = (float*)d_out;

  // workspace carve-up (256B aligned)
  char* p = (char*)d_ws;
  auto take = [&](size_t bytes) { char* r = p; p += (bytes + 255) & ~(size_t)255; return r; };
  _Float16* W1 = (_Float16*)take((size_t)N1 * Dm * 2);          // [up_l ; up_r]         f16
  _Float16* W2 = (_Float16*)take((size_t)N2P * Pp * 2);         // [fused ; skip ; 0pad] f16
  _Float16* W3 = (_Float16*)take((size_t)Dm * HID * 2);         // down                  f16
  _Float16* XN = (_Float16*)take((size_t)Mrows * Dm * 2);       // ln(x)                 f16
  float*    Y1 = (float*)   take((size_t)Mrows * N1 * 4);       // [xl ; r]              f32
  _Float16* XC = (_Float16*)take((size_t)Mrows * Pp * 2);       // silu(conv(xl))        f16
  float*    Z  = (float*)   take((size_t)Mrows * N2P * 4);      // [z ; skip ; pad]      f32
  float*    H  = (float*)   take((size_t)Mrows * HID * 4);      // gated head out        f32
  _Float16* U  = (_Float16*)take((size_t)Mrows * HID * 2);      // LN(h)+skip etc        f16

  // 1) weight converts (cheap, deterministic every call)
  {
    int n;
    n = Pp * Dm;    cvt_f32_to_f16<<<(n + 255) / 256, 256, 0, stream>>>(up_l_w, W1, n);
    n = HID * Dm;   cvt_f32_to_f16<<<(n + 255) / 256, 256, 0, stream>>>(up_r_w, W1 + (size_t)Pp * Dm, n);
    n = ZB * Pp;    cvt_f32_to_f16<<<(n + 255) / 256, 256, 0, stream>>>(fused_w, W2, n);
    n = HID * Pp;   cvt_f32_to_f16<<<(n + 255) / 256, 256, 0, stream>>>(skip_w, W2 + (size_t)ZB * Pp, n);
    n = (N2P - N2) * Pp; zero_f16<<<(n + 255) / 256, 256, 0, stream>>>(W2 + (size_t)N2 * Pp, n);
    n = Dm * HID;   cvt_f32_to_f16<<<(n + 255) / 256, 256, 0, stream>>>(down_w, W3, n);
  }

  // 2) LayerNorm -> f16
  ln_kernel<<<Mrows, 256, 0, stream>>>(x, ln_w, ln_b, XN);

  // 3) GEMM A: Y1[M, 2304] = XN * W1^T     (xl | r)
  {
    int blocks = (Mrows / 64) * (N1 / 32) / 8;
    gemm_wmma_f16<<<blocks, 256, 0, stream>>>(XN, W1, Y1, nullptr, 0, nullptr, Mrows, N1, Dm);
  }

  // 4) feature-dim causal conv + SiLU -> f16
  conv_silu_kernel<<<(Mrows * Pp) / 256, 256, 0, stream>>>(Y1, conv_w, XC);

  // 5) GEMM B: Z[M, 3872] = XC * W2^T + bias(first 3088)   (z | skip | pad)
  {
    int blocks = (Mrows / 64) * (N2P / 32) / 8;
    gemm_wmma_f16<<<blocks, 256, 0, stream>>>(XC, W2, Z, fused_b, ZB, nullptr, Mrows, N2P, Pp);
  }

  // 6) sequential matrix-state recurrence: 64 independent (b,h) chains
  recurrence_kernel<<<Bb * NH, 96, 0, stream>>>(Z, H);

  // 7) hid-LN + skip + silu(r) -> f16
  post_kernel<<<Mrows, 256, 0, stream>>>(H, Z, Y1, hid_w, hid_b, U);

  // 8) GEMM C: out[M, 768] = U * W3^T + x (residual)
  {
    int blocks = (Mrows / 64) * (Dm / 32) / 8;
    gemm_wmma_f16<<<blocks, 256, 0, stream>>>(U, W3, out, nullptr, 0, x, Mrows, Dm, HID);
  }
}
